// GraphTransformerLayer_14534169329722
// MI455X (gfx1250) — matrix operand order repaired and checked by reading, not compiled, not run on hardware
//
#include <hip/hip_runtime.h>
#include <cstdint>
#include <cstddef>

// Problem constants (from reference)
#define N_NODES 50000
#define DIM     128
#define NHEAD   8
#define DHEAD   16
#define NB      8
#define MAXN    6250
#define NEDGE   400000
#define NGTOK   8      // global tokens per batch
#define DFF     256

typedef float  v8f  __attribute__((ext_vector_type(8)));
typedef __bf16 v8bf __attribute__((ext_vector_type(8)));
typedef __bf16 v16bf __attribute__((ext_vector_type(16)));

__device__ __forceinline__ v16bf cat16(v8bf lo, v8bf hi) {
  return __builtin_shufflevector(lo, hi, 0,1,2,3,4,5,6,7,8,9,10,11,12,13,14,15);
}

__device__ __forceinline__ float wave_sum(float v) {
#pragma unroll
  for (int m = 16; m >= 1; m >>= 1) v += __shfl_xor(v, m, 32);
  return v;
}

// ---- float <-> ordered-uint key for atomic float max --------------------
__device__ __forceinline__ unsigned fkey(float f) {
  unsigned u = __float_as_uint(f);
  return (u & 0x80000000u) ? ~u : (u | 0x80000000u);
}
__device__ __forceinline__ float funkey(unsigned k) {
  unsigned u = (k & 0x80000000u) ? (k & 0x7fffffffu) : ~k;
  return __uint_as_float(u);
}

// =========================================================================
// Generic bf16 GEMM:  C[M x NOUT] = A[M x K](bf16) * W[NOUT x K]^T + bias
// 256 threads = 8 waves; each block does 128 rows; WMMA 16x16x32 bf16.
// K / NOUT / RELU / WB are compile-time: immediate-offset stores, no branches.
// =========================================================================
template <int K, int NOUT, bool RELU, bool WB>
__global__ __launch_bounds__(256)
void gemm_bf16_nt(const __bf16* __restrict__ A, const __bf16* __restrict__ W,
                  const float* __restrict__ bias, float* __restrict__ C,
                  __bf16* __restrict__ Cb, int M) {
  constexpr int NK = K / 32;
  __shared__ __bf16 lw[64 * K];

  const int tid  = threadIdx.x;
  const int wid  = tid >> 5;
  const int lane = tid & 31;
  const int rowBase = blockIdx.x * 128 + wid * 16;
  const int halfSel = (lane < 16) ? 0 : 1;
  const int rl = lane & 15;
  const bool fullTile = (blockIdx.x + 1) * 128 <= M;

  int arow = rowBase + rl;
  if (arow > M - 1) arow = M - 1;
  const __bf16* aptr = A + (size_t)arow * K;

  // A fragments: lane<16 holds K {0..7,16..23}+32*kf ; lane>=16 holds {8..15,24..31}
  v16bf afrag[NK];
#pragma unroll
  for (int kf = 0; kf < NK; ++kf) {
    const int k0 = kf * 32 + halfSel * 8;
    v8bf lo = *(const v8bf*)(aptr + k0);
    v8bf hi = *(const v8bf*)(aptr + k0 + 16);
    afrag[kf] = cat16(lo, hi);
  }

#pragma unroll 1
  for (int n0 = 0; n0 < NOUT; n0 += 64) {
    __syncthreads();
#pragma unroll
    for (int ld = 0; ld < (64 * K) / (8 * 256); ++ld) {
      const int idx = ld * 256 + tid;
      *(v8bf*)(lw + idx * 8) = *(const v8bf*)(W + (size_t)n0 * K + idx * 8);
    }
    // prefetch next weight tile into cache while we compute on this one
    if (n0 + 64 < NOUT) {
      __builtin_prefetch(W + (size_t)(n0 + 64) * K + (size_t)tid * (K / 4), 0, 1);
    }
    __syncthreads();

#pragma unroll
    for (int ns = 0; ns < 4; ++ns) {
      // load all B fragments for this n-subtile first (overlapped DS loads),
      // then run the WMMA chain back-to-back.
      v16bf bfr[NK];
#pragma unroll
      for (int kf = 0; kf < NK; ++kf) {
        const __bf16* bp = lw + (size_t)(ns * 16 + rl) * K + kf * 32 + halfSel * 8;
        v8bf lo = *(const v8bf*)bp;
        v8bf hi = *(const v8bf*)(bp + 16);
        bfr[kf] = cat16(lo, hi);
      }
      v8f acc = {};
#pragma unroll
      for (int kf = 0; kf < NK; ++kf) {
        acc = __builtin_amdgcn_wmma_f32_16x16x32_bf16(
            false, afrag[kf], false, bfr[kf], (short)0, acc, false, false);
      }

      const int n = n0 + ns * 16 + rl;
      const float bz = bias[n];
      float vals[8];
#pragma unroll
      for (int r = 0; r < 8; ++r) {
        float v = acc[r] + bz;
        if (RELU) v = fmaxf(v, 0.0f);
        vals[r] = v;
      }
      const int row0 = rowBase + halfSel * 8;
      if (fullTile) {
        float* cp = C + (size_t)row0 * NOUT + n;
#pragma unroll
        for (int r = 0; r < 8; ++r) cp[(size_t)r * NOUT] = vals[r];
        if (WB) {
          __bf16* cbp = Cb + (size_t)row0 * NOUT + n;
#pragma unroll
          for (int r = 0; r < 8; ++r) cbp[(size_t)r * NOUT] = (__bf16)vals[r];
        }
      } else {
#pragma unroll
        for (int r = 0; r < 8; ++r) {
          const int row = row0 + r;
          if (row < M) {
            const size_t idx = (size_t)row * NOUT + n;
            C[idx] = vals[r];
            if (WB) Cb[idx] = (__bf16)vals[r];
          }
        }
      }
    }
  }
}

// =========================================================================
// small helpers
// =========================================================================
__global__ void cvt_bf16_kernel(const float* __restrict__ s, __bf16* __restrict__ d, int n) {
  for (int i = blockIdx.x * blockDim.x + threadIdx.x; i < n; i += gridDim.x * blockDim.x)
    d[i] = (__bf16)s[i];
}
__global__ void copy_f32_kernel(const float* __restrict__ s, float* __restrict__ d, int n) {
  for (int i = blockIdx.x * blockDim.x + threadIdx.x; i < n; i += gridDim.x * blockDim.x)
    d[i] = s[i];
}

// =========================================================================
// Edge attention (tconv): pass 1  — logits + atomic segment max
// one wave per edge; lane covers 4 dims; head = lane>>2
// =========================================================================
__global__ __launch_bounds__(256)
void edge_logits_kernel(const int* __restrict__ src, const int* __restrict__ dst,
                        const float* __restrict__ qkvs, float* __restrict__ alog,
                        unsigned* __restrict__ mkey) {
  const int wid = threadIdx.x >> 5, lane = threadIdx.x & 31;
  const int e = blockIdx.x * 8 + wid;
  if (e >= NEDGE) return;
  const int s = src[e], d = dst[e];
  const int c = lane * 4;
  const float* q = qkvs + (size_t)d * 512 + c;        // q cols 0..127
  const float* k = qkvs + (size_t)s * 512 + 128 + c;  // k cols 128..255
  float p = 0.f;
#pragma unroll
  for (int j = 0; j < 4; ++j) p += q[j] * k[j];
  p += __shfl_xor(p, 1, 32);
  p += __shfl_xor(p, 2, 32);              // per-head (quad) sum
  if ((lane & 3) == 0) {
    const int h = lane >> 2;
    const float logit = p * 0.25f;        // 1/sqrt(16)
    alog[(size_t)e * 8 + h] = logit;
    atomicMax(mkey + (size_t)d * 8 + h, fkey(logit));
  }
}

// pass 2: a = exp(l - m[dst]); atomic segment sum
__global__ __launch_bounds__(256)
void edge_exp_kernel(const int* __restrict__ dst, float* __restrict__ alog,
                     const unsigned* __restrict__ mkey, float* __restrict__ ssum) {
  const long long gid = (long long)blockIdx.x * blockDim.x + threadIdx.x;
  if (gid >= (long long)NEDGE * 8) return;
  const int e = (int)(gid >> 3), h = (int)(gid & 7);
  const int d = dst[e];
  const unsigned k = mkey[(size_t)d * 8 + h];
  const float m = k ? funkey(k) : 0.0f;   // no-edge segments -> 0 (matches isfinite guard)
  const float a = __expf(alog[gid] - m);
  alog[gid] = a;
  atomicAdd(ssum + (size_t)d * 8 + h, a);
}

// pass 3: out[dst] += v[src] * alpha ; one wave per edge
__global__ __launch_bounds__(256)
void edge_accum_kernel(const int* __restrict__ src, const int* __restrict__ dst,
                       const float* __restrict__ qkvs, const float* __restrict__ alog,
                       const float* __restrict__ ssum, float* __restrict__ outagg) {
  const int wid = threadIdx.x >> 5, lane = threadIdx.x & 31;
  const int e = blockIdx.x * 8 + wid;
  if (e >= NEDGE) return;
  const int s = src[e], d = dst[e];
  const int h = lane >> 2;
  const float alpha = alog[(size_t)e * 8 + h] / (ssum[(size_t)d * 8 + h] + 1e-16f);
  const int c = lane * 4;
  const float* v = qkvs + (size_t)s * 512 + 256 + c;  // v cols 256..383
  float* o = outagg + (size_t)d * 128 + c;
#pragma unroll
  for (int j = 0; j < 4; ++j) atomicAdd(o + j, v[j] * alpha);
}

// pass 4: beta gate + residual + LN -> x1 (f32 + bf16); one wave per node
__global__ __launch_bounds__(256)
void node_combine_ln_kernel(const float* __restrict__ x, const float* __restrict__ qkvs,
                            const float* __restrict__ outagg, const float* __restrict__ Wbeta,
                            float* __restrict__ x1, __bf16* __restrict__ x1b) {
  const int wid = threadIdx.x >> 5, lane = threadIdx.x & 31;
  const int i = blockIdx.x * 8 + wid;
  if (i >= N_NODES) return;
  const int c = lane * 4;
  float ov[4], xr[4];
  float dotp = 0.f;
#pragma unroll
  for (int j = 0; j < 4; ++j) {
    ov[j] = outagg[(size_t)i * 128 + c + j];
    xr[j] = qkvs[(size_t)i * 512 + 384 + c + j];       // skip cols 384..511
    dotp += Wbeta[c + j] * ov[j] + Wbeta[128 + c + j] * xr[j]
          + Wbeta[256 + c + j] * (ov[j] - xr[j]);
  }
  const float beta = 1.0f / (1.0f + __expf(-wave_sum(dotp)));
  float y[4];
  float s = 0.f;
#pragma unroll
  for (int j = 0; j < 4; ++j) {
    const float hl = beta * xr[j] + (1.0f - beta) * ov[j];
    y[j] = x[(size_t)i * 128 + c + j] + hl;
    s += y[j];
  }
  const float mean = wave_sum(s) * (1.0f / 128.0f);
  float vs = 0.f;
#pragma unroll
  for (int j = 0; j < 4; ++j) { const float dd = y[j] - mean; vs += dd * dd; }
  const float rs = rsqrtf(wave_sum(vs) * (1.0f / 128.0f) + 1e-5f);
#pragma unroll
  for (int j = 0; j < 4; ++j) {
    const float o = (y[j] - mean) * rs;
    x1[(size_t)i * 128 + c + j] = o;
    x1b[(size_t)i * 128 + c + j] = (__bf16)o;
  }
}

// residual + LN ; one wave per row
__global__ __launch_bounds__(256)
void ln_add_kernel(const float* __restrict__ a, const float* __restrict__ b,
                   float* __restrict__ out, __bf16* __restrict__ outb, int M) {
  const int wid = threadIdx.x >> 5, lane = threadIdx.x & 31;
  const int row = blockIdx.x * 8 + wid;
  if (row >= M) return;
  const int c = lane * 4;
  float y[4];
  float s = 0.f;
#pragma unroll
  for (int j = 0; j < 4; ++j) {
    y[j] = a[(size_t)row * 128 + c + j] + b[(size_t)row * 128 + c + j];
    s += y[j];
  }
  const float mean = wave_sum(s) * (1.0f / 128.0f);
  float vs = 0.f;
#pragma unroll
  for (int j = 0; j < 4; ++j) { const float dd = y[j] - mean; vs += dd * dd; }
  const float rs = rsqrtf(wave_sum(vs) * (1.0f / 128.0f) + 1e-5f);
#pragma unroll
  for (int j = 0; j < 4; ++j) {
    const float o = (y[j] - mean) * rs;
    out[(size_t)row * 128 + c + j] = o;
    if (outb) outb[(size_t)row * 128 + c + j] = (__bf16)o;
  }
}

// =========================================================================
// MHA1: global tokens attend to nodes
// =========================================================================
__global__ __launch_bounds__(128)
void m1_q_kernel(const float* __restrict__ gh, const float* __restrict__ ng_in_w,
                 const float* __restrict__ ng_in_b, float* __restrict__ Qg) {
  __shared__ float row[128];
  const int g = blockIdx.x, j = threadIdx.x;
  row[j] = gh[(size_t)g * 128 + j];
  __syncthreads();
  float acc = ng_in_b[j];
  const float* w = ng_in_w + (size_t)j * 128;   // wq rows 0..127
  for (int k = 0; k < 128; ++k) acc += row[k] * w[k];
  Qg[(size_t)g * 128 + j] = acc;
}

// flash-style attention: one block per (b,h,t) over MAXN keys
__global__ __launch_bounds__(256)
void m1_attn_kernel(const float* __restrict__ Qg, const float* __restrict__ kv,
                    float* __restrict__ O1, float* __restrict__ m1m, float* __restrict__ m1s) {
  const int bi = blockIdx.x;
  const int b = bi >> 6, h = (bi >> 3) & 7, t = bi & 7;
  const int tid = threadIdx.x;
  __shared__ float qsh[16];
  __shared__ float red_m[256], red_s[256], red_a[256 * 16];
  if (tid < 16) qsh[tid] = Qg[((size_t)(b * NGTOK + t)) * 128 + h * 16 + tid];
  __syncthreads();

  float mloc = -1e30f, sloc = 0.f, acc[16];
#pragma unroll
  for (int d = 0; d < 16; ++d) acc[d] = 0.f;

  for (int n = tid; n < MAXN; n += 256) {
    const float* krow = kv + ((size_t)(b * MAXN + n)) * 256 + h * 16;
    float l = 0.f;
#pragma unroll
    for (int d = 0; d < 16; ++d) l += qsh[d] * krow[d];
    l *= 0.25f;
    const float mn = fmaxf(mloc, l);
    const float sc = __expf(mloc - mn);
    const float w  = __expf(l - mn);
    sloc = sloc * sc + w;
    const float* vrow = krow + 128;
#pragma unroll
    for (int d = 0; d < 16; ++d) acc[d] = acc[d] * sc + w * vrow[d];
    mloc = mn;
  }
  red_m[tid] = mloc; red_s[tid] = sloc;
#pragma unroll
  for (int d = 0; d < 16; ++d) red_a[tid * 16 + d] = acc[d];
  __syncthreads();
  for (int off = 128; off >= 1; off >>= 1) {
    if (tid < off) {
      const float m2 = red_m[tid + off];
      const float mn = fmaxf(red_m[tid], m2);
      const float e1 = __expf(red_m[tid] - mn), e2 = __expf(m2 - mn);
      red_s[tid] = red_s[tid] * e1 + red_s[tid + off] * e2;
#pragma unroll
      for (int d = 0; d < 16; ++d)
        red_a[tid * 16 + d] = red_a[tid * 16 + d] * e1 + red_a[(tid + off) * 16 + d] * e2;
      red_m[tid] = mn;
    }
    __syncthreads();
  }
  if (tid < 16)
    O1[((size_t)(b * NGTOK + t)) * 128 + h * 16 + tid] = red_a[tid] / red_s[0];
  if (tid == 0) {
    const int idx = (b * NGTOK + t) * NHEAD + h;
    m1m[idx] = red_m[0];
    m1s[idx] = red_s[0];
  }
}

// out proj + residual + LN for 64 tokens -> d_out + ws copy
__global__ __launch_bounds__(128)
void m1_out_kernel(const float* __restrict__ O1, const float* __restrict__ gh,
                   const float* __restrict__ ow, const float* __restrict__ ob,
                   float* __restrict__ gh_out, float* __restrict__ gh_ws) {
  __shared__ float row[128];
  __shared__ float part[4];
  const int g = blockIdx.x, j = threadIdx.x;
  const int wid = j >> 5;
  row[j] = O1[(size_t)g * 128 + j];
  __syncthreads();
  float acc = ob[j];
  const float* w = ow + (size_t)j * 128;
  for (int k = 0; k < 128; ++k) acc += row[k] * w[k];
  float y = gh[(size_t)g * 128 + j] + acc;
  float s = wave_sum(y);
  if ((j & 31) == 0) part[wid] = s;
  __syncthreads();
  const float mean = (part[0] + part[1] + part[2] + part[3]) * (1.0f / 128.0f);
  const float dd = y - mean;
  float vsum = wave_sum(dd * dd);
  if ((j & 31) == 0) part[wid] = vsum;
  __syncthreads();
  const float var = (part[0] + part[1] + part[2] + part[3]) * (1.0f / 128.0f);
  const float o = dd * rsqrtf(var + 1e-5f);
  gh_out[(size_t)g * 128 + j] = o;
  gh_ws[(size_t)g * 128 + j] = o;
}

// mean-over-heads attention probs for local tokens (t'=4..7)
__global__ __launch_bounds__(256)
void m1_p_kernel(const float* __restrict__ Qg, const float* __restrict__ kv,
                 const float* __restrict__ m1m, const float* __restrict__ m1s,
                 float* __restrict__ p) {
  const int bi = blockIdx.x;          // 0..31
  const int b = bi >> 2, tt = bi & 3;
  const int tok = 4 + tt;
  __shared__ float qsh[128], msh[8], ssh[8];
  const int tid = threadIdx.x;
  if (tid < 128) qsh[tid] = Qg[((size_t)(b * NGTOK + tok)) * 128 + tid];
  if (tid < 8) {
    msh[tid] = m1m[(b * NGTOK + tok) * NHEAD + tid];
    ssh[tid] = m1s[(b * NGTOK + tok) * NHEAD + tid];
  }
  __syncthreads();
  for (int n = tid; n < MAXN; n += 256) {
    const float* krow = kv + ((size_t)(b * MAXN + n)) * 256;
    float acc = 0.f;
#pragma unroll
    for (int h = 0; h < 8; ++h) {
      float l = 0.f;
#pragma unroll
      for (int d = 0; d < 16; ++d) l += qsh[h * 16 + d] * krow[h * 16 + d];
      l *= 0.25f;
      acc += __expf(l - msh[h]) / ssh[h];
    }
    p[((size_t)(b * 4 + tt)) * MAXN + n] = acc * (1.0f / 8.0f);
  }
}

__device__ __forceinline__ float bsum256(float v, float* buf, int tid) {
  __syncthreads();
  buf[tid] = v;
  __syncthreads();
  for (int off = 128; off >= 1; off >>= 1) {
    if (tid < off) buf[tid] += buf[tid + off];
    __syncthreads();
  }
  return buf[0];
}

// diversity regularizer -> single scalar (one block)
__global__ __launch_bounds__(256)
void m1_reg_kernel(const float* __restrict__ p, float* __restrict__ reg_out) {
  __shared__ float buf[256];
  const int tid = threadIdx.x;
  float regsum = 0.f;
  for (int b = 0; b < NB; ++b) {
    float R[4] = {0, 0, 0, 0};
    float G[16];
#pragma unroll
    for (int i = 0; i < 16; ++i) G[i] = 0.f;
    for (int n = tid; n < MAXN; n += 256) {
      float pv[4];
#pragma unroll
      for (int k = 0; k < 4; ++k) pv[k] = p[((size_t)(b * 4 + k)) * MAXN + n];
#pragma unroll
      for (int k = 0; k < 4; ++k) {
        R[k] += pv[k];
#pragma unroll
        for (int m = 0; m < 4; ++m) G[k * 4 + m] += pv[k] * pv[m];
      }
    }
    float Rr[4], Gr[16];
#pragma unroll
    for (int k = 0; k < 4; ++k) Rr[k] = bsum256(R[k], buf, tid);
#pragma unroll
    for (int i = 0; i < 16; ++i) Gr[i] = bsum256(G[i], buf, tid);
    if (tid == 0) {
      float tot = 0.f, tr = 0.f;
      for (int k = 0; k < 4; ++k)
        for (int m = 0; m < 4; ++m) {
          const float g = Gr[k * 4 + m] / ((Rr[k] + 1e-8f) * (Rr[m] + 1e-8f));
          tot += g;
          if (k == m) tr += g;
        }
      regsum += (tot - tr) * (1.0f / 12.0f);   // Kt^2 - Kt = 12
    }
  }
  if (tid == 0) reg_out[0] = regsum * (1.0f / (float)NB);
}

// =========================================================================
// MHA2: nodes attend to 8 global tokens
// =========================================================================
__global__ __launch_bounds__(256)
void m2_kvg_kernel(const float* __restrict__ gh, const float* __restrict__ gn_in_w,
                   const float* __restrict__ gn_in_b, float* __restrict__ Kg,
                   float* __restrict__ Vg) {
  __shared__ float row[128];
  const int g = blockIdx.x, tid = threadIdx.x;
  if (tid < 128) row[tid] = gh[(size_t)g * 128 + tid];
  __syncthreads();
  if (tid < 128) {
    const float* w = gn_in_w + (size_t)(128 + tid) * 128;     // wk
    float acc = gn_in_b[128 + tid];
    for (int k = 0; k < 128; ++k) acc += row[k] * w[k];
    Kg[(size_t)g * 128 + tid] = acc;
  } else {
    const int j = tid - 128;
    const float* w = gn_in_w + (size_t)(256 + j) * 128;       // wv
    float acc = gn_in_b[256 + j];
    for (int k = 0; k < 128; ++k) acc += row[k] * w[k];
    Vg[(size_t)g * 128 + j] = acc;
  }
}

__global__ __launch_bounds__(256)
void m2_attn_kernel(const float* __restrict__ qn, const float* __restrict__ Kg,
                    const float* __restrict__ Vg, float* __restrict__ O2,
                    __bf16* __restrict__ O2b) {
  const long long gid = (long long)blockIdx.x * blockDim.x + threadIdx.x;
  if (gid >= (long long)N_NODES * NHEAD) return;
  const int n = (int)(gid >> 3), h = (int)(gid & 7);
  const int b = n / MAXN;
  const float* q = qn + (size_t)n * 128 + h * 16;
  float l[8];
  float mx = -1e30f;
#pragma unroll
  for (int t = 0; t < 8; ++t) {
    const float* kr = Kg + ((size_t)(b * NGTOK + t)) * 128 + h * 16;
    float d = 0.f;
#pragma unroll
    for (int j = 0; j < 16; ++j) d += q[j] * kr[j];
    l[t] = d * 0.25f;
    mx = fmaxf(mx, l[t]);
  }
  float s = 0.f;
#pragma unroll
  for (int t = 0; t < 8; ++t) { l[t] = __expf(l[t] - mx); s += l[t]; }
  const float inv = 1.0f / s;
  float o[16];
#pragma unroll
  for (int j = 0; j < 16; ++j) o[j] = 0.f;
#pragma unroll
  for (int t = 0; t < 8; ++t) {
    const float w = l[t] * inv;
    const float* vr = Vg + ((size_t)(b * NGTOK + t)) * 128 + h * 16;
#pragma unroll
    for (int j = 0; j < 16; ++j) o[j] += w * vr[j];
  }
#pragma unroll
  for (int j = 0; j < 16; ++j) {
    O2[(size_t)n * 128 + h * 16 + j] = o[j];
    O2b[(size_t)n * 128 + h * 16 + j] = (__bf16)o[j];
  }
}

// =========================================================================
// Host-side orchestration
// =========================================================================
extern "C" void kernel_launch(void* const* d_in, const int* in_sizes, int n_in,
                              void* d_out, int out_size, void* d_ws, size_t ws_size,
                              hipStream_t stream) {
  (void)in_sizes; (void)n_in; (void)out_size; (void)ws_size;
  const float* x        = (const float*)d_in[0];
  const int*   eidx     = (const int*)d_in[1];
  const float* gh_in    = (const float*)d_in[2];
  const float* Wq       = (const float*)d_in[6];
  const float* bq       = (const float*)d_in[7];
  const float* Wk       = (const float*)d_in[8];
  const float* bk       = (const float*)d_in[9];
  const float* Wv       = (const float*)d_in[10];
  const float* bv       = (const float*)d_in[11];
  const float* Wskip    = (const float*)d_in[12];
  const float* bskip    = (const float*)d_in[13];
  const float* Wbeta    = (const float*)d_in[14];
  const float* ng_in_w  = (const float*)d_in[15];
  const float* ng_in_b  = (const float*)d_in[16];
  const float* ng_out_w = (const float*)d_in[17];
  const float* ng_out_b = (const float*)d_in[18];
  const float* gn_in_w  = (const float*)d_in[19];
  const float* gn_in_b  = (const float*)d_in[20];
  const float* gn_out_w = (const float*)d_in[21];
  const float* gn_out_b = (const float*)d_in[22];
  const float* W1       = (const float*)d_in[23];
  const float* b1       = (const float*)d_in[24];
  const float* W2       = (const float*)d_in[25];
  const float* b2       = (const float*)d_in[26];

  const int* src = eidx;
  const int* dst = eidx + NEDGE;

  char* ws = (char*)d_ws;
  size_t off = 0;
  auto take = [&](size_t bytes) { size_t r = off; off += (bytes + 255) & ~(size_t)255; return r; };

  const size_t o_wcat   = take(512 * 128 * 2);          // bf16 Wq|Wk|Wv|Wskip
  const size_t o_ngkv   = take(256 * 128 * 2);          // bf16 ng wk|wv
  const size_t o_gnq    = take(128 * 128 * 2);          // bf16 gn wq
  const size_t o_gnout  = take(128 * 128 * 2);
  const size_t o_w1     = take(256 * 128 * 2);
  const size_t o_w2     = take(128 * 256 * 2);
  const size_t o_bcat   = take(512 * 4);
  const size_t o_xb     = take((size_t)N_NODES * 128 * 2);
  const size_t o_qkvs   = take((size_t)N_NODES * 512 * 4);  // later: kv | qn | O2
  const size_t o_alog   = take((size_t)NEDGE * 8 * 4);      // later: O2 bf16
  const size_t o_mkey   = take((size_t)N_NODES * 8 * 4);
  const size_t o_s      = take((size_t)N_NODES * 8 * 4);
  const size_t o_outagg = take((size_t)N_NODES * 128 * 4);  // later: proj2
  const size_t o_x1     = take((size_t)N_NODES * 128 * 4);  // later: ff1 bf16
  const size_t o_x1b    = take((size_t)N_NODES * 128 * 2);  // later: x2 bf16
  const size_t o_x2     = take((size_t)N_NODES * 128 * 4);
  const size_t o_Qg     = take(64 * 128 * 4);
  const size_t o_O1     = take(64 * 128 * 4);
  const size_t o_m1m    = take(64 * 8 * 4);
  const size_t o_m1s    = take(64 * 8 * 4);
  const size_t o_gh     = take(64 * 128 * 4);
  const size_t o_kvg    = take(2 * 64 * 128 * 4);
  const size_t o_p      = take((size_t)NB * 4 * MAXN * 4);

  // region aliases (sequential reuse)
  float*  qkvs   = (float*)(ws + o_qkvs);
  float*  kv     = (float*)(ws + o_qkvs);                                   // N x 256
  float*  qn     = (float*)(ws + o_qkvs + (size_t)N_NODES * 256 * 4);       // N x 128
  float*  O2     = (float*)(ws + o_qkvs + (size_t)N_NODES * 384 * 4);       // N x 128
  float*  ff1    = (float*)(ws + o_qkvs);                                   // N x 256
  float*  ff2    = qn;                                                      // N x 128
  float*  alog   = (float*)(ws + o_alog);
  __bf16* O2b    = (__bf16*)(ws + o_alog);
  float*  outagg = (float*)(ws + o_outagg);
  float*  proj2  = (float*)(ws + o_outagg);
  float*  x1     = (float*)(ws + o_x1);
  __bf16* ff1b   = (__bf16*)(ws + o_x1);
  __bf16* x1b    = (__bf16*)(ws + o_x1b);
  __bf16* x2b    = (__bf16*)(ws + o_x1b);
  float*  x2     = (float*)(ws + o_x2);

  float* out_x   = (float*)d_out;
  float* out_gh  = (float*)d_out + (size_t)N_NODES * 128;
  float* out_reg = out_gh + 64 * 128;

  // zero accumulators (captured as memset nodes)
  hipMemsetAsync(ws + o_mkey,   0, (size_t)N_NODES * 8 * 4, stream);
  hipMemsetAsync(ws + o_s,      0, (size_t)N_NODES * 8 * 4, stream);
  hipMemsetAsync(ws + o_outagg, 0, (size_t)N_NODES * 128 * 4, stream);

  auto cvt = [&](const float* s_, size_t dof, int n) {
    int blocks = (n + 255) / 256; if (blocks > 1024) blocks = 1024;
    cvt_bf16_kernel<<<blocks, 256, 0, stream>>>(s_, (__bf16*)(ws + dof), n);
  };
  // weights -> bf16
  cvt(Wq,    o_wcat,               128 * 128);
  cvt(Wk,    o_wcat + 16384 * 2,   128 * 128);
  cvt(Wv,    o_wcat + 32768 * 2,   128 * 128);
  cvt(Wskip, o_wcat + 49152 * 2,   128 * 128);
  cvt(ng_in_w + 128 * 128, o_ngkv, 256 * 128);
  cvt(gn_in_w, o_gnq,              128 * 128);
  cvt(gn_out_w, o_gnout,           128 * 128);
  cvt(W1, o_w1,                    256 * 128);
  cvt(W2, o_w2,                    128 * 256);
  cvt(x,  o_xb, N_NODES * 128);
  // bias concat for qkvs
  copy_f32_kernel<<<1, 128, 0, stream>>>(bq,    (float*)(ws + o_bcat) + 0,   128);
  copy_f32_kernel<<<1, 128, 0, stream>>>(bk,    (float*)(ws + o_bcat) + 128, 128);
  copy_f32_kernel<<<1, 128, 0, stream>>>(bv,    (float*)(ws + o_bcat) + 256, 128);
  copy_f32_kernel<<<1, 128, 0, stream>>>(bskip, (float*)(ws + o_bcat) + 384, 128);

  const int gemmBlocks = (N_NODES + 127) / 128;

  // ---- stage 1: q|k|v|skip projection (WMMA) ----
  gemm_bf16_nt<128, 512, false, false><<<gemmBlocks, 256, 0, stream>>>(
      (const __bf16*)(ws + o_xb), (const __bf16*)(ws + o_wcat),
      (const float*)(ws + o_bcat), qkvs, nullptr, N_NODES);

  // ---- stage 2: edge attention ----
  edge_logits_kernel<<<(NEDGE + 7) / 8, 256, 0, stream>>>(src, dst, qkvs, alog,
                                                          (unsigned*)(ws + o_mkey));
  edge_exp_kernel<<<(NEDGE * 8 + 255) / 256, 256, 0, stream>>>(dst, alog,
                                                               (const unsigned*)(ws + o_mkey),
                                                               (float*)(ws + o_s));
  edge_accum_kernel<<<(NEDGE + 7) / 8, 256, 0, stream>>>(src, dst, qkvs, alog,
                                                         (const float*)(ws + o_s), outagg);
  node_combine_ln_kernel<<<(N_NODES + 7) / 8, 256, 0, stream>>>(x, qkvs, outagg, Wbeta,
                                                                x1, x1b);

  // ---- stage 3: MHA1 (global <- nodes) ----
  m1_q_kernel<<<64, 128, 0, stream>>>(gh_in, ng_in_w, ng_in_b, (float*)(ws + o_Qg));
  gemm_bf16_nt<128, 256, false, false><<<gemmBlocks, 256, 0, stream>>>(
      x1b, (const __bf16*)(ws + o_ngkv), ng_in_b + 128, kv, nullptr, N_NODES);
  m1_attn_kernel<<<512, 256, 0, stream>>>((const float*)(ws + o_Qg), kv,
                                          (float*)(ws + o_O1), (float*)(ws + o_m1m),
                                          (float*)(ws + o_m1s));
  m1_out_kernel<<<64, 128, 0, stream>>>((const float*)(ws + o_O1), gh_in, ng_out_w,
                                        ng_out_b, out_gh, (float*)(ws + o_gh));
  m1_p_kernel<<<32, 256, 0, stream>>>((const float*)(ws + o_Qg), kv,
                                      (const float*)(ws + o_m1m), (const float*)(ws + o_m1s),
                                      (float*)(ws + o_p));
  m1_reg_kernel<<<1, 256, 0, stream>>>((const float*)(ws + o_p), out_reg);

  // ---- stage 4: MHA2 (nodes <- global) ----
  m2_kvg_kernel<<<64, 256, 0, stream>>>((const float*)(ws + o_gh), gn_in_w, gn_in_b,
                                        (float*)(ws + o_kvg),
                                        (float*)(ws + o_kvg) + 64 * 128);
  gemm_bf16_nt<128, 128, false, false><<<gemmBlocks, 256, 0, stream>>>(
      x1b, (const __bf16*)(ws + o_gnq), gn_in_b, qn, nullptr, N_NODES);
  m2_attn_kernel<<<((long long)N_NODES * NHEAD + 255) / 256, 256, 0, stream>>>(
      qn, (const float*)(ws + o_kvg), (const float*)(ws + o_kvg) + 64 * 128, O2, O2b);
  gemm_bf16_nt<128, 128, false, false><<<gemmBlocks, 256, 0, stream>>>(
      O2b, (const __bf16*)(ws + o_gnout), gn_out_b, proj2, nullptr, N_NODES);
  ln_add_kernel<<<(N_NODES + 7) / 8, 256, 0, stream>>>(x1, proj2, x2, x2b, N_NODES);

  // ---- stage 5: FFN + final LN -> d_out ----
  gemm_bf16_nt<128, 256, true, true><<<gemmBlocks, 256, 0, stream>>>(
      x2b, (const __bf16*)(ws + o_w1), b1, ff1, ff1b, N_NODES);
  gemm_bf16_nt<256, 128, false, false><<<gemmBlocks, 256, 0, stream>>>(
      ff1b, (const __bf16*)(ws + o_w2), b2, ff2, nullptr, N_NODES);
  ln_add_kernel<<<(N_NODES + 7) / 8, 256, 0, stream>>>(x2, ff2, out_x, nullptr, N_NODES);
}
